// MLA_14998025797769
// MI455X (gfx1250) — matrix operand order repaired and checked by reading, not compiled, not run on hardware
//
#include <hip/hip_runtime.h>
#include <hip/hip_bf16.h>

// ---------------------------------------------------------------------------
// MLA forward for MI455X (gfx1250): all GEMMs via v_wmma_f32_16x16x32_bf16,
// flash-style attention (no materialized 256MB score tensor).
// Round 3: ping-pong double buffering (no register-rotation movs) in both the
// GEMM K-loop and the attention key-block loop; global_prefetch_b8 streaming.
// ---------------------------------------------------------------------------

typedef __attribute__((ext_vector_type(16))) __bf16 v16bf;
typedef __attribute__((ext_vector_type(8)))  __bf16 v8bf;
typedef __attribute__((ext_vector_type(8)))  float  v8f;

#define BS     16
#define SEQ    512
#define DMODEL 1024
#define DOWN   256
#define UPD    512
#define NH     16
#define RHD    64
#define QKD    96      // 32 (compressed head) + 64 (rope)
#define VHD    32
#define TOKENS (BS * SEQ)   // 8192

// ---------------------------------------------------------------------------
// WMMA fragment helpers (layouts per CDNA5 ISA 7.12.2, wave32)
// ---------------------------------------------------------------------------

// A fragment: 16x32 bf16, row-major source. `p` = &A[row_base + lane%16][kk].
__device__ __forceinline__ v16bf load_a_frag(const __bf16* p, int half) {
  v8bf lo = *reinterpret_cast<const v8bf*>(p + half * 8);
  v8bf hi = *reinterpret_cast<const v8bf*>(p + 16 + half * 8);
  v16bf a;
#pragma unroll
  for (int i = 0; i < 8; ++i) { a[i] = lo[i]; a[i + 8] = hi[i]; }
  return a;
}

// B fragment: 32x16 bf16 from an N-major (transposed) source.
__device__ __forceinline__ v16bf load_b_frag(const __bf16* p, int half) {
  v8bf lo = *reinterpret_cast<const v8bf*>(p + half * 8);
  v8bf hi = *reinterpret_cast<const v8bf*>(p + 16 + half * 8);
  v16bf b;
#pragma unroll
  for (int i = 0; i < 8; ++i) { b[i] = lo[i]; b[i + 8] = hi[i]; }
  return b;
}

__device__ __forceinline__ v8f wmma_bf16(v16bf a, v16bf b, v8f c) {
  return __builtin_amdgcn_wmma_f32_16x16x32_bf16(
      false, a, false, b, (short)0, c, false, false);
}

// ---------------------------------------------------------------------------
// Elementwise prep kernels
// ---------------------------------------------------------------------------

__global__ void mla_cvt_f32_bf16(const float* __restrict__ x,
                                 __bf16* __restrict__ y, int n) {
  int i = blockIdx.x * blockDim.x + threadIdx.x;
  if (i < n) y[i] = (__bf16)x[i];
}

// W[K,N] f32 -> Wt[N,K] bf16
__global__ void mla_wtrans(const float* __restrict__ W,
                           __bf16* __restrict__ Wt, int K, int N) {
  int i = blockIdx.x * blockDim.x + threadIdx.x;
  if (i >= K * N) return;
  int n = i % N, k = i / N;
  Wt[(size_t)n * K + k] = (__bf16)W[i];
}

// ---------------------------------------------------------------------------
// GEMM: C[M,N] = A[M,K](bf16) * Bt[N,K](bf16)^T + bias.
// One wave -> 32x64 tile: 2 M-subtiles x 4 N-subtiles = 8 accumulator chains.
// K-loop unrolled 2x into explicit ping-pong fragment sets (no rotation movs);
// each WMMA group runs under the outstanding loads of the next K-step.
// M % 32 == 0, N % 64 == 0, K % 64 == 0.
// ---------------------------------------------------------------------------
__global__ void __launch_bounds__(256)
mla_gemm_bf16(const __bf16* __restrict__ A, const __bf16* __restrict__ Bt,
              const float* __restrict__ bias, void* __restrict__ Cout,
              int M, int N, int K, int out_bf16) {
  int wid = blockIdx.x * (blockDim.x >> 5) + (threadIdx.x >> 5);
  int tiles_n = N >> 6;
  int tiles   = (M >> 5) * tiles_n;
  if (wid >= tiles) return;          // whole-wave uniform exit (EXEC stays full)
  int m0 = (wid / tiles_n) << 5;
  int n0 = (wid % tiles_n) << 6;
  int lane = threadIdx.x & 31, half = lane >> 4, l16 = lane & 15;

  v8f acc[2][4];
#pragma unroll
  for (int t = 0; t < 4; ++t) {
    float bv = bias[n0 + t * 16 + l16];
#pragma unroll
    for (int r = 0; r < 8; ++r) { acc[0][t][r] = bv; acc[1][t][r] = bv; }
  }

  const __bf16* arow0 = A + (size_t)(m0 + l16) * K;
  const __bf16* arow1 = A + (size_t)(m0 + 16 + l16) * K;
  const __bf16* brow[4];
#pragma unroll
  for (int t = 0; t < 4; ++t) brow[t] = Bt + (size_t)(n0 + t * 16 + l16) * K;

  // ping set (X) at kk, pong set (Y) at kk+32
  v16bf a0X, a1X, bX[4], a0Y, a1Y, bY[4];
  a0X = load_a_frag(arow0, half);
  a1X = load_a_frag(arow1, half);
#pragma unroll
  for (int t = 0; t < 4; ++t) bX[t] = load_b_frag(brow[t], half);

  for (int kk = 0; kk < K; kk += 64) {
    if (kk + 256 < K) {   // streaming prefetch (global_prefetch_b8)
      __builtin_prefetch(arow0 + kk + 256, 0, 1);
      __builtin_prefetch(brow[0] + kk + 256, 0, 1);
    }

    // load pong set (kk+32) -- always in-bounds since K % 64 == 0
    a0Y = load_a_frag(arow0 + kk + 32, half);
    a1Y = load_a_frag(arow1 + kk + 32, half);
#pragma unroll
    for (int t = 0; t < 4; ++t) bY[t] = load_b_frag(brow[t] + kk + 32, half);

    // consume ping set
#pragma unroll
    for (int t = 0; t < 4; ++t) acc[0][t] = wmma_bf16(a0X, bX[t], acc[0][t]);
#pragma unroll
    for (int t = 0; t < 4; ++t) acc[1][t] = wmma_bf16(a1X, bX[t], acc[1][t]);

    // load next ping set (kk+64, clamped to 0 on last trip)
    int kn = (kk + 64 < K) ? kk + 64 : 0;
    a0X = load_a_frag(arow0 + kn, half);
    a1X = load_a_frag(arow1 + kn, half);
#pragma unroll
    for (int t = 0; t < 4; ++t) bX[t] = load_b_frag(brow[t] + kn, half);

    // consume pong set
#pragma unroll
    for (int t = 0; t < 4; ++t) acc[0][t] = wmma_bf16(a0Y, bY[t], acc[0][t]);
#pragma unroll
    for (int t = 0; t < 4; ++t) acc[1][t] = wmma_bf16(a1Y, bY[t], acc[1][t]);
  }

#pragma unroll
  for (int mt = 0; mt < 2; ++mt)
#pragma unroll
    for (int t = 0; t < 4; ++t)
#pragma unroll
      for (int r = 0; r < 8; ++r) {
        int row = m0 + mt * 16 + r + half * 8;
        int col = n0 + t * 16 + l16;
        if (out_bf16)
          ((__bf16*)Cout)[(size_t)row * N + col] = (__bf16)acc[mt][t][r];
        else
          ((float*)Cout)[(size_t)row * N + col] = acc[mt][t][r];
      }
}

// ---------------------------------------------------------------------------
// Assemble q/k (concat compressed head + RoPE'd rope-head) into
// [bs, nh, seq, 96] bf16. Faithful to the reference's swapped cos/sin cache.
// ---------------------------------------------------------------------------
__global__ void mla_build_qk(const __bf16* __restrict__ q_c,
                             const __bf16* __restrict__ qr_pre,
                             const __bf16* __restrict__ k_c,
                             const __bf16* __restrict__ kr_pre,
                             __bf16* __restrict__ qout,
                             __bf16* __restrict__ kout) {
  int idx = blockIdx.x * blockDim.x + threadIdx.x;
  const int total = BS * NH * SEQ * QKD;
  if (idx >= total) return;
  int d = idx % QKD;
  int s = (idx / QKD) % SEQ;
  int h = (idx / (QKD * SEQ)) % NH;
  int b = idx / (QKD * SEQ * NH);
  size_t tok = (size_t)b * SEQ + s;

  float qv, kv;
  if (d < VHD) {
    qv = (float)q_c[tok * UPD + h * VHD + d];
    kv = (float)k_c[tok * UPD + h * VHD + d];
  } else {
    int dr = d - VHD;
    int i  = dr >> 1;
    // theta = 10000^(-2i/64); cos_pos holds sin(pe), sin_pos holds cos(pe)
    float theta = __expf(-(float)(2 * i) * (9.210340371976184f / 64.0f));
    float pe = (float)s * theta;
    float sp = __sinf(pe), cp = __cosf(pe);

    float xq  = (float)qr_pre[tok * (NH * RHD) + h * RHD + dr];
    float xqn = (float)qr_pre[tok * (NH * RHD) + h * RHD + (dr ^ 1)];
    float x2q = (dr & 1) ? xqn : -xqn;
    qv = xq * sp + x2q * cp;

    float xk  = (float)kr_pre[tok * RHD + dr];
    float xkn = (float)kr_pre[tok * RHD + (dr ^ 1)];
    float x2k = (dr & 1) ? xkn : -xkn;
    kv = xk * sp + x2k * cp;
  }
  size_t o = (((size_t)b * NH + h) * SEQ + s) * QKD + d;
  qout[o] = (__bf16)qv;
  kout[o] = (__bf16)kv;
}

// v_c [tok, nh*32] bf16 -> vt [bs, nh, 32, seq] bf16 (N-major for WMMA B)
__global__ void mla_build_vt(const __bf16* __restrict__ v_c,
                             __bf16* __restrict__ vt) {
  int idx = blockIdx.x * blockDim.x + threadIdx.x;
  const int total = BS * NH * VHD * SEQ;
  if (idx >= total) return;
  int s = idx % SEQ;
  int d = (idx / SEQ) % VHD;
  int h = (idx / (SEQ * VHD)) % NH;
  int b = idx / (SEQ * VHD * NH);
  vt[idx] = v_c[((size_t)b * SEQ + s) * UPD + h * VHD + d];
}

// ---------------------------------------------------------------------------
// Flash attention: block = 8 waves; wave -> 16 query rows; block -> (b, h,
// 128-query slab). Online softmax; scores never touch HBM. Scale = 1/16
// (sqrt(64)+sqrt(64)); mask==0 -> -1e9 before scaling (matches reference).
// Key-block loop is ping-pong double buffered: while block i runs its score
// WMMAs + softmax + P*V, block i+1's K/V/mask data is already in flight.
// ---------------------------------------------------------------------------

struct KBlk {
  v16bf bk0[3], bk1[3];   // key fragments (two 16-key tiles x 3 K-chunks)
  v16bf bv0, bv1;         // value fragments (two 16-col tiles)
  int   mv0[8], mv1[8];   // mask words
};

__device__ __forceinline__ void load_kblk(KBlk& s, const __bf16* kp,
                                          const __bf16* vp,
                                          const int* __restrict__ mask,
                                          int q0, int kb, int half, int l16) {
#pragma unroll
  for (int c = 0; c < 3; ++c) {
    s.bk0[c] = load_b_frag(kp + (size_t)(kb + l16) * QKD + c * 32, half);
    s.bk1[c] = load_b_frag(kp + (size_t)(kb + 16 + l16) * QKD + c * 32, half);
  }
  s.bv0 = load_b_frag(vp + (size_t)l16 * SEQ + kb, half);
  s.bv1 = load_b_frag(vp + (size_t)(16 + l16) * SEQ + kb, half);
#pragma unroll
  for (int r = 0; r < 8; ++r) {
    int qrow = q0 + r + half * 8;
    s.mv0[r] = mask[qrow * SEQ + kb + l16];
    s.mv1[r] = mask[qrow * SEQ + kb + 16 + l16];
  }
}

__device__ __forceinline__ void proc_kblk(const KBlk& s, const v16bf aq[3],
                                          float mrow[8], float lrow[8],
                                          v8f& o0, v8f& o1, __bf16* lp,
                                          int half, int l16) {
  // ---- scores for 32 keys: two 16x16 tiles, K=96 ----
  v8f s0 = {}, s1 = {};
#pragma unroll
  for (int c = 0; c < 3; ++c) {
    s0 = wmma_bf16(aq[c], s.bk0[c], s0);
    s1 = wmma_bf16(aq[c], s.bk1[c], s1);
  }

  // ---- mask + scale + online softmax (row = r + half*8 within tile) ----
  float p0[8], p1[8];
#pragma unroll
  for (int r = 0; r < 8; ++r) {
    float x0 = (s.mv0[r] ? s0[r] : -1e9f) * 0.0625f;
    float x1 = (s.mv1[r] ? s1[r] : -1e9f) * 0.0625f;

    float mx = fmaxf(x0, x1);
#pragma unroll
    for (int d = 1; d < 16; d <<= 1) mx = fmaxf(mx, __shfl_xor(mx, d, 32));
    float mnew = fmaxf(mrow[r], mx);
    float corr = __expf(mrow[r] - mnew);
    float e0 = __expf(x0 - mnew);
    float e1 = __expf(x1 - mnew);
    float rs = e0 + e1;
#pragma unroll
    for (int d = 1; d < 16; d <<= 1) rs += __shfl_xor(rs, d, 32);
    lrow[r] = lrow[r] * corr + rs;
    mrow[r] = mnew;
    o0[r] *= corr;
    o1[r] *= corr;
    p0[r] = e0;
    p1[r] = e1;
  }

  // ---- re-shuffle P (C layout) -> A fragment layout via LDS ----
#pragma unroll
  for (int r = 0; r < 8; ++r) {
    int row = r + half * 8;
    lp[row * 32 + l16]      = (__bf16)p0[r];
    lp[row * 32 + 16 + l16] = (__bf16)p1[r];
  }
  asm volatile("s_wait_dscnt 0" ::: "memory");
  v16bf pa = load_a_frag(lp + l16 * 32, half);

  // ---- O += P(16x32) x V(32x32): V fragments already in flight ----
  o0 = wmma_bf16(pa, s.bv0, o0);
  o1 = wmma_bf16(pa, s.bv1, o1);
}

__global__ void __launch_bounds__(256)
mla_attention(const __bf16* __restrict__ q, const __bf16* __restrict__ k,
              const __bf16* __restrict__ vt, const int* __restrict__ mask,
              __bf16* __restrict__ attn_out) {
  __shared__ __bf16 lds_p[8 * 16 * 32];   // per-wave 16x32 P staging

  int qblk = blockIdx.x, h = blockIdx.y, b = blockIdx.z;
  int wave = threadIdx.x >> 5, lane = threadIdx.x & 31;
  int half = lane >> 4, l16 = lane & 15;
  int q0 = qblk * 128 + wave * 16;

  const __bf16* qp = q  + ((size_t)(b * NH + h) * SEQ) * QKD;
  const __bf16* kp = k  + ((size_t)(b * NH + h) * SEQ) * QKD;
  const __bf16* vp = vt + ((size_t)(b * NH + h) * VHD) * SEQ;
  __bf16* lp = lds_p + wave * 16 * 32;

  // Q fragments for the full d=96 (3 x K=32 chunks), kept resident.
  v16bf aq[3];
#pragma unroll
  for (int c = 0; c < 3; ++c)
    aq[c] = load_a_frag(qp + (size_t)(q0 + l16) * QKD + c * 32, half);

  float mrow[8], lrow[8];
  v8f o0 = {}, o1 = {};
#pragma unroll
  for (int r = 0; r < 8; ++r) { mrow[r] = -1e30f; lrow[r] = 0.0f; }

  KBlk X, Y;
  load_kblk(X, kp, vp, mask, q0, 0, half, l16);

  for (int kb = 0; kb < SEQ; kb += 64) {
    // pong block (kb+32) in flight while ping block computes
    load_kblk(Y, kp, vp, mask, q0, kb + 32, half, l16);
    proc_kblk(X, aq, mrow, lrow, o0, o1, lp, half, l16);

    // next ping block (kb+64, clamped on last trip) in flight during pong
    int kn = (kb + 64 < SEQ) ? kb + 64 : 0;
    load_kblk(X, kp, vp, mask, q0, kn, half, l16);
    proc_kblk(Y, aq, mrow, lrow, o0, o1, lp, half, l16);
  }

  // normalize and emit [bs, seq, nh*32] bf16 for the final FC
#pragma unroll
  for (int r = 0; r < 8; ++r) {
    float inv = 1.0f / lrow[r];
    int qrow = q0 + r + half * 8;
    size_t base = ((size_t)b * SEQ + qrow) * (NH * VHD) + h * VHD;
    attn_out[base + l16]      = (__bf16)(o0[r] * inv);
    attn_out[base + 16 + l16] = (__bf16)(o1[r] * inv);
  }
}

// ---------------------------------------------------------------------------
// Host orchestration
// ---------------------------------------------------------------------------

static inline void launch_gemm(const __bf16* A, const __bf16* Bt,
                               const float* bias, void* C, int M, int N, int K,
                               int out_bf16, hipStream_t s) {
  int waves = (M / 32) * (N / 64);
  int blocks = (waves + 7) / 8;
  hipLaunchKernelGGL(mla_gemm_bf16, dim3(blocks), dim3(256), 0, s,
                     A, Bt, bias, C, M, N, K, out_bf16);
}

extern "C" void kernel_launch(void* const* d_in, const int* in_sizes, int n_in,
                              void* d_out, int out_size, void* d_ws, size_t ws_size,
                              hipStream_t stream) {
  const float* h     = (const float*)d_in[0];
  const int*   mask  = (const int*)  d_in[1];
  const float* w_dkv = (const float*)d_in[2];  const float* b_dkv = (const float*)d_in[3];
  const float* w_uk  = (const float*)d_in[4];  const float* b_uk  = (const float*)d_in[5];
  const float* w_uv  = (const float*)d_in[6];  const float* b_uv  = (const float*)d_in[7];
  const float* w_dq  = (const float*)d_in[8];  const float* b_dq  = (const float*)d_in[9];
  const float* w_uq  = (const float*)d_in[10]; const float* b_uq  = (const float*)d_in[11];
  const float* w_qr  = (const float*)d_in[12]; const float* b_qr  = (const float*)d_in[13];
  const float* w_kr  = (const float*)d_in[14]; const float* b_kr  = (const float*)d_in[15];
  const float* w_fc  = (const float*)d_in[16]; const float* b_fc  = (const float*)d_in[17];
  float* out = (float*)d_out;

  // --- workspace layout ---
  char* ws = (char*)d_ws;
  size_t off = 0;
  auto alloc = [&](size_t elems) { void* p = ws + off; off += ((elems * 2 + 255) & ~(size_t)255); return p; };
  __bf16* h_bf    = (__bf16*)alloc((size_t)TOKENS * DMODEL);
  __bf16* c_kv    = (__bf16*)alloc((size_t)TOKENS * DOWN);
  __bf16* c_q     = (__bf16*)alloc((size_t)TOKENS * DOWN);
  __bf16* kr_pre  = (__bf16*)alloc((size_t)TOKENS * RHD);
  __bf16* k_c     = (__bf16*)alloc((size_t)TOKENS * UPD);
  __bf16* v_c     = (__bf16*)alloc((size_t)TOKENS * UPD);
  __bf16* q_c     = (__bf16*)alloc((size_t)TOKENS * UPD);
  __bf16* qr_pre  = (__bf16*)alloc((size_t)TOKENS * NH * RHD);
  __bf16* q_bf    = (__bf16*)alloc((size_t)BS * NH * SEQ * QKD);
  __bf16* k_bf    = (__bf16*)alloc((size_t)BS * NH * SEQ * QKD);
  __bf16* vt_bf   = (__bf16*)alloc((size_t)BS * NH * VHD * SEQ);
  __bf16* attn_bf = (__bf16*)alloc((size_t)TOKENS * UPD);
  __bf16* wt_dkv  = (__bf16*)alloc((size_t)DMODEL * DOWN);
  __bf16* wt_dq   = (__bf16*)alloc((size_t)DMODEL * DOWN);
  __bf16* wt_kr   = (__bf16*)alloc((size_t)DMODEL * RHD);
  __bf16* wt_uk   = (__bf16*)alloc((size_t)DOWN * UPD);
  __bf16* wt_uv   = (__bf16*)alloc((size_t)DOWN * UPD);
  __bf16* wt_uq   = (__bf16*)alloc((size_t)DOWN * UPD);
  __bf16* wt_qr   = (__bf16*)alloc((size_t)DOWN * NH * RHD);
  __bf16* wt_fc   = (__bf16*)alloc((size_t)UPD * DMODEL);

  const int T = 256;
  auto grid1 = [&](int n) { return dim3((n + T - 1) / T); };

  // 1) precision conversion + weight transposes
  int nh_elems = TOKENS * DMODEL;
  hipLaunchKernelGGL(mla_cvt_f32_bf16, grid1(nh_elems), dim3(T), 0, stream, h, h_bf, nh_elems);
  hipLaunchKernelGGL(mla_wtrans, grid1(DMODEL * DOWN), dim3(T), 0, stream, w_dkv, wt_dkv, DMODEL, DOWN);
  hipLaunchKernelGGL(mla_wtrans, grid1(DMODEL * DOWN), dim3(T), 0, stream, w_dq,  wt_dq,  DMODEL, DOWN);
  hipLaunchKernelGGL(mla_wtrans, grid1(DMODEL * RHD),  dim3(T), 0, stream, w_kr,  wt_kr,  DMODEL, RHD);
  hipLaunchKernelGGL(mla_wtrans, grid1(DOWN * UPD),    dim3(T), 0, stream, w_uk,  wt_uk,  DOWN, UPD);
  hipLaunchKernelGGL(mla_wtrans, grid1(DOWN * UPD),    dim3(T), 0, stream, w_uv,  wt_uv,  DOWN, UPD);
  hipLaunchKernelGGL(mla_wtrans, grid1(DOWN * UPD),    dim3(T), 0, stream, w_uq,  wt_uq,  DOWN, UPD);
  hipLaunchKernelGGL(mla_wtrans, grid1(DOWN * NH*RHD), dim3(T), 0, stream, w_qr,  wt_qr,  DOWN, NH * RHD);
  hipLaunchKernelGGL(mla_wtrans, grid1(UPD * DMODEL),  dim3(T), 0, stream, w_fc,  wt_fc,  UPD, DMODEL);

  // 2) down-projections + rope-key projection (from h)
  launch_gemm(h_bf, wt_dkv, b_dkv, c_kv,   TOKENS, DOWN, DMODEL, 1, stream);
  launch_gemm(h_bf, wt_dq,  b_dq,  c_q,    TOKENS, DOWN, DMODEL, 1, stream);
  launch_gemm(h_bf, wt_kr,  b_kr,  kr_pre, TOKENS, RHD,  DMODEL, 1, stream);

  // 3) up-projections
  launch_gemm(c_kv, wt_uk, b_uk, k_c,    TOKENS, UPD,      DOWN, 1, stream);
  launch_gemm(c_kv, wt_uv, b_uv, v_c,    TOKENS, UPD,      DOWN, 1, stream);
  launch_gemm(c_q,  wt_uq, b_uq, q_c,    TOKENS, UPD,      DOWN, 1, stream);
  launch_gemm(c_q,  wt_qr, b_qr, qr_pre, TOKENS, NH * RHD, DOWN, 1, stream);

  // 4) RoPE + layout assembly
  hipLaunchKernelGGL(mla_build_qk, grid1(BS * NH * SEQ * QKD), dim3(T), 0, stream,
                     q_c, qr_pre, k_c, kr_pre, q_bf, k_bf);
  hipLaunchKernelGGL(mla_build_vt, grid1(BS * NH * VHD * SEQ), dim3(T), 0, stream,
                     v_c, vt_bf);

  // 5) flash attention
  hipLaunchKernelGGL(mla_attention, dim3(SEQ / 128, NH, BS), dim3(256), 0, stream,
                     q_bf, k_bf, vt_bf, mask, attn_bf);

  // 6) output projection -> f32 result
  launch_gemm(attn_bf, wt_fc, b_fc, out, TOKENS, DMODEL, UPD, 0, stream);
}
